// Moe_layer_46583215292722
// MI455X (gfx1250) — matrix operand order repaired and checked by reading, not compiled, not run on hardware
//
#include <hip/hip_runtime.h>
#include <hip/hip_bf16.h>
#include <math.h>

#define NTOK 8192
#define DDIM 1024
#define FDIM 4096
#define NEXP 8

typedef __attribute__((ext_vector_type(16))) __bf16          v16bf;
typedef __attribute__((ext_vector_type(8)))  float           v8f;
typedef __attribute__((ext_vector_type(8)))  unsigned short  ushort8;

union Frag {
  ushort8        h[2];
  unsigned short u[16];
  v16bf          v;
};

// fp32 -> bf16 round-to-nearest-even (bit-level)
__device__ __forceinline__ unsigned short f2bf(float f) {
  unsigned int u = __builtin_bit_cast(unsigned int, f);
  u += 0x7FFFu + ((u >> 16) & 1u);
  return (unsigned short)(u >> 16);
}

// ---------------- Gating: logits, top-2, bucket append ----------------
__global__ __launch_bounds__(256) void moe_gate(const float* __restrict__ x,
                                                const float* __restrict__ Wg,
                                                int*   __restrict__ cnt,
                                                int*   __restrict__ btok,
                                                float* __restrict__ bgate) {
  const int wave = threadIdx.x >> 5;
  const int lane = threadIdx.x & 31;
  const int n = blockIdx.x * 8 + wave;

  float acc[NEXP];
#pragma unroll
  for (int e = 0; e < NEXP; ++e) acc[e] = 0.f;

  const float* xr = x + (size_t)n * DDIM;
  for (int d = lane; d < DDIM; d += 32) {
    const float xv = xr[d];
    const float* wr = Wg + (size_t)d * NEXP;
#pragma unroll
    for (int e = 0; e < NEXP; ++e) acc[e] += xv * wr[e];
  }
#pragma unroll
  for (int e = 0; e < NEXP; ++e) {
#pragma unroll
    for (int m = 16; m >= 1; m >>= 1) acc[e] += __shfl_xor(acc[e], m, 32);
  }

  if (lane == 0) {
    float mx = acc[0];
#pragma unroll
    for (int e = 1; e < NEXP; ++e) mx = fmaxf(mx, acc[e]);
    float ex[NEXP];
#pragma unroll
    for (int e = 0; e < NEXP; ++e) ex[e] = __expf(acc[e] - mx);

    int e0 = 0; float v0 = ex[0];
#pragma unroll
    for (int e = 1; e < NEXP; ++e) if (ex[e] > v0) { v0 = ex[e]; e0 = e; }
    int e1 = -1; float v1 = -1.f;
#pragma unroll
    for (int e = 0; e < NEXP; ++e) if (e != e0 && ex[e] > v1) { v1 = ex[e]; e1 = e; }

    const float inv = 1.f / (v0 + v1);
    int p0 = atomicAdd(&cnt[e0], 1);
    btok [e0 * NTOK + p0] = n;
    bgate[e0 * NTOK + p0] = v0 * inv;
    int p1 = atomicAdd(&cnt[e1], 1);
    btok [e1 * NTOK + p1] = n;
    bgate[e1 * NTOK + p1] = v1 * inv;
  }
}

// -------- Weight convert + swizzle into WMMA B-fragment lane layout --------
// src: [E][KD][ND] fp32 row-major (W1: KD=D,ND=F ; W2: KD=F,ND=D).
// dst: per 32(K)x16(N) tile, 32 lanes x 16 bf16 contiguous (1 KB):
//   lane l covers N = n0*16 + (l&15), K = k0*32 + (l>>4)*16 + j, j=0..15.
__global__ __launch_bounds__(256) void swizzle_w(const float* __restrict__ src,
                                                 unsigned short* __restrict__ dst,
                                                 int KD, int ND) {
  const size_t t = (size_t)blockIdx.x * 256 + threadIdx.x;  // = tile*32 + lane
  const int lane = (int)(t & 31);
  const size_t tile = t >> 5;                 // (e*(ND/16)+n0)*(KD/32)+k0
  const int kt = KD >> 5;
  const int k0 = (int)(tile % kt);
  const size_t en = tile / kt;                // e*(ND/16)+n0
  const int nt = ND >> 4;
  const int n0 = (int)(en % nt);
  const int e  = (int)(en / nt);
  const int n = n0 * 16 + (lane & 15);
  const int kb = k0 * 32 + ((lane >> 4) << 4);

  const float* s = src + ((size_t)e * KD + kb) * ND + n;
  Frag f;
#pragma unroll
  for (int j = 0; j < 16; ++j) f.u[j] = f2bf(s[(size_t)j * ND]);
  ushort8* d = (ushort8*)(dst + t * 16);
  d[0] = f.h[0];
  d[1] = f.h[1];
}

// ---------------- Fused expert FFN: GEMM1 + relu (h in LDS) + GEMM2 ----------------
__global__ __launch_bounds__(256) void moe_expert(
    const float* __restrict__ x,
    const unsigned short* __restrict__ w1s,   // swizzled bf16 [E][F/16][D/32] tiles
    const float* __restrict__ b1,
    const unsigned short* __restrict__ w2s,   // swizzled bf16 [E][D/16][F/32] tiles
    const float* __restrict__ b2,
    const int*   __restrict__ cnt,
    const int*   __restrict__ btok,
    const float* __restrict__ bgate,
    float* __restrict__ out) {
  const int e = blockIdx.y;
  const int count = cnt[e];
  const int row0 = (int)blockIdx.x * 16;
  if (row0 >= count) return;

  extern __shared__ char smem[];
  unsigned short* xs = (unsigned short*)smem;   // 16 x 1024 bf16 (32 KB)
  unsigned short* hs = xs + 16 * DDIM;          // 16 x 4096 bf16 (128 KB)
  int*   toks  = (int*)(hs + 16 * FDIM);
  float* gates = (float*)(toks + 16);

  const int tid = threadIdx.x;
  if (tid < 16) {
    const int idx = row0 + tid;
    toks [tid] = (idx < count) ? btok [e * NTOK + idx] : -1;
    gates[tid] = (idx < count) ? bgate[e * NTOK + idx] : 0.f;
  }
  __syncthreads();

  for (int i = tid; i < 16 * DDIM; i += 256) {
    const int r = i >> 10, c = i & (DDIM - 1);
    const int t = toks[r];
    xs[i] = f2bf((t >= 0) ? x[(size_t)t * DDIM + c] : 0.f);
  }
  __syncthreads();

  const int wave = tid >> 5;
  const int lane = tid & 31;
  const int lcol = lane & 15;   // A: row M ; B/C: column N
  const int hi   = lane >> 4;

  // ---- GEMM1: h = relu(xs @ W1[e] + b1[e]); 256 tiles, 32/wave, groups of 4 ----
  // Uniform base pointer + 32-bit lane offsets so B loads lower to global_load_b128.
  {
    const ushort8* w1v =
        (const ushort8*)(w1s + (size_t)e * (FDIM / 16) * (DDIM / 32) * 512);
    for (int g = 0; g < 8; ++g) {
      v8f acc[4] = {};
      unsigned off[4];  // in ushort8 (16-byte) units; 64 per 32x16 K-step tile
#pragma unroll
      for (int u = 0; u < 4; ++u) {
        const int tile = wave + 8 * (4 * g + u);
        off[u] = (unsigned)tile * (DDIM / 32) * 64 + (unsigned)lane * 2;
      }
      for (int kk = 0; kk < DDIM; kk += 32) {
        Frag a;
        const unsigned short* ap = xs + lcol * DDIM + kk + hi * 16;
        a.h[0] = *(const ushort8*)ap;
        a.h[1] = *(const ushort8*)(ap + 8);
#pragma unroll
        for (int u = 0; u < 4; ++u) {
          Frag b;
          b.h[0] = w1v[off[u]];
          b.h[1] = w1v[off[u] + 1];
          __builtin_prefetch(w1v + off[u] + 64, 0, 1);
          acc[u] = __builtin_amdgcn_wmma_f32_16x16x32_bf16(false, a.v, false, b.v,
                                                           (short)0, acc[u], false, false);
          off[u] += 64;
        }
      }
#pragma unroll
      for (int u = 0; u < 4; ++u) {
        const int tile = wave + 8 * (4 * g + u);
        const int f = tile * 16 + lcol;
        const float bias = b1[e * FDIM + f];
#pragma unroll
        for (int r = 0; r < 8; ++r) {
          const float v = acc[u][r] + bias;
          hs[(r + hi * 8) * FDIM + f] = f2bf(v > 0.f ? v : 0.f);
        }
      }
    }
  }
  __syncthreads();

  // ---- GEMM2: y[tok] += gate * (hs @ W2[e] + b2[e]); 64 tiles, 8/wave, groups of 4 ----
  {
    const ushort8* w2v =
        (const ushort8*)(w2s + (size_t)e * (DDIM / 16) * (FDIM / 32) * 512);
    for (int g = 0; g < 2; ++g) {
      v8f acc[4] = {};
      unsigned off[4];
#pragma unroll
      for (int u = 0; u < 4; ++u) {
        const int tile = wave + 8 * (4 * g + u);
        off[u] = (unsigned)tile * (FDIM / 32) * 64 + (unsigned)lane * 2;
      }
      for (int kk = 0; kk < FDIM; kk += 32) {
        Frag a;
        const unsigned short* ap = hs + lcol * FDIM + kk + hi * 16;
        a.h[0] = *(const ushort8*)ap;
        a.h[1] = *(const ushort8*)(ap + 8);
#pragma unroll
        for (int u = 0; u < 4; ++u) {
          Frag b;
          b.h[0] = w2v[off[u]];
          b.h[1] = w2v[off[u] + 1];
          __builtin_prefetch(w2v + off[u] + 64, 0, 1);
          acc[u] = __builtin_amdgcn_wmma_f32_16x16x32_bf16(false, a.v, false, b.v,
                                                           (short)0, acc[u], false, false);
          off[u] += 64;
        }
      }
#pragma unroll
      for (int u = 0; u < 4; ++u) {
        const int tile = wave + 8 * (4 * g + u);
        const int dcol = tile * 16 + lcol;
        const float bias = b2[e * DDIM + dcol];
#pragma unroll
        for (int r = 0; r < 8; ++r) {
          const int m = r + hi * 8;
          const int t = toks[m];
          if (t >= 0)
            atomicAdd(out + (size_t)t * DDIM + dcol, (acc[u][r] + bias) * gates[m]);
        }
      }
    }
  }
}

extern "C" void kernel_launch(void* const* d_in, const int* in_sizes, int n_in,
                              void* d_out, int out_size, void* d_ws, size_t ws_size,
                              hipStream_t stream) {
  const float* x  = (const float*)d_in[0];
  const float* Wg = (const float*)d_in[1];
  const float* W1 = (const float*)d_in[2];
  const float* b1 = (const float*)d_in[3];
  const float* W2 = (const float*)d_in[4];
  const float* b2 = (const float*)d_in[5];
  float* out = (float*)d_out;

  char* ws = (char*)d_ws;
  int*   cnt   = (int*)ws;                                     // 8 counters (pad 64B)
  int*   btok  = (int*)(ws + 64);                              // E x N token ids
  float* bgate = (float*)(ws + 64 + (size_t)NEXP * NTOK * 4);  // E x N gates
  unsigned short* w1s = (unsigned short*)(ws + (1u << 20));                    // 64 MB
  unsigned short* w2s = (unsigned short*)(ws + (1u << 20) +
                                          (size_t)NEXP * DDIM * FDIM * 2);     // 64 MB

  hipMemsetAsync(cnt, 0, 64, stream);
  hipMemsetAsync(out, 0, (size_t)NTOK * DDIM * sizeof(float), stream);

  // Weight convert+swizzle: 2,097,152 threads each (8192 blocks of 256)
  const int swblocks = (int)(((size_t)NEXP * (FDIM / 16) * (DDIM / 32) * 32) / 256);
  swizzle_w<<<swblocks, 256, 0, stream>>>(W1, w1s, DDIM, FDIM);
  swizzle_w<<<swblocks, 256, 0, stream>>>(W2, w2s, FDIM, DDIM);

  moe_gate<<<NTOK / 8, 256, 0, stream>>>(x, Wg, cnt, btok, bgate);

  const size_t lds = (size_t)(16 * DDIM + 16 * FDIM) * sizeof(unsigned short)
                   + 16 * sizeof(int) + 16 * sizeof(float);    // ~160 KB (<= 320 KB WGP)
  dim3 grid(NTOK / 16, NEXP);
  moe_expert<<<grid, 256, lds, stream>>>(x, w1s, b1, w2s, b2, cnt, btok, bgate, out);
}
